// TemporalKplanesOffsetfields_57080115364240
// MI455X (gfx1250) — matrix operand order repaired and checked by reading, not compiled, not run on hardware
//
#include <hip/hip_runtime.h>

typedef __attribute__((ext_vector_type(16))) _Float16 v16h;
typedef __attribute__((ext_vector_type(8)))  _Float16 v8h;
typedef __attribute__((ext_vector_type(8)))  float    v8f;

#define NPTS   128          // points per block (4 wave32s)
#define FEAT   64
#define KDIM   128          // 2*FEAT
#define HID    64
#define LDS_FK 136          // padded feature row (halves): 272B, 16B-aligned, bank-rotating
#define LDS_WK 136          // padded transposed-w0 row (halves per n-column)
#define LDS_HT 132          // padded transposed-h row (floats per n-column)

// ---- LDS layout (overlayed, barrier-separated phases) ----
// [0      , 34816) : sFeat f16 [128][136]   -- later reused as hT f32 [64][132] (33792B)
// [34816  , 52224) : sW0t f16 [64][136]     (w0 transposed: [n][k])
// [52224  , 53264) : sW1 f32[192], sB1 f32[3], sB0 f32[64]
#define OFF_W0  34816
#define OFF_SM  52224
#define SMEM_BYTES 53264

// A fragment: 16x32 f16 (MxK). Lane l: M = rowBase+(l&15); halves =
// K[kLo..kLo+15], kLo = (l<16)?0:16.  Two aligned 16B LDS loads.
__device__ __forceinline__ v16h ldA(const _Float16* feat, int rowBase, int kBase, int lane) {
    const _Float16* p = feat + (rowBase + (lane & 15)) * LDS_FK + kBase + ((lane < 16) ? 0 : 16);
    v8h lo = *(const v8h*)(p);
    v8h hi = *(const v8h*)(p + 8);
    return __builtin_shufflevector(lo, hi, 0,1,2,3,4,5,6,7,8,9,10,11,12,13,14,15);
}

// B fragment: 32x16 f16 (KxN). Lane l: N = nBase+(l&15); halves = K[k0..k0+15],
// k0 = kBase + ((l<16)?0:16).  w0 stored transposed [n][k] -> 32 contiguous bytes.
__device__ __forceinline__ v16h ldB(const _Float16* wt, int kBase, int nBase, int lane) {
    const _Float16* p = wt + (nBase + (lane & 15)) * LDS_WK + kBase + ((lane < 16) ? 0 : 16);
    v8h lo = *(const v8h*)(p);
    v8h hi = *(const v8h*)(p + 8);
    return __builtin_shufflevector(lo, hi, 0,1,2,3,4,5,6,7,8,9,10,11,12,13,14,15);
}

__device__ __forceinline__ unsigned pk2(float a, float b) {
    union { _Float16 h[2]; unsigned u; } x;
    x.h[0] = (_Float16)a; x.h[1] = (_Float16)b;
    return x.u;
}

__global__ void __launch_bounds__(NPTS)
kplanes_mlp_kernel(const float* __restrict__ inp, const float* __restrict__ aabb,
                   const float* __restrict__ tp0, const float* __restrict__ tp1,
                   const float* __restrict__ tp2, const float* __restrict__ tp3,
                   const float* __restrict__ tp4, const float* __restrict__ tp5,
                   const float* __restrict__ sp0, const float* __restrict__ sp1,
                   const float* __restrict__ sp2,
                   const float* __restrict__ w0, const float* __restrict__ b0,
                   const float* __restrict__ w1, const float* __restrict__ b1,
                   float* __restrict__ out, int n)
{
    __shared__ __align__(16) char smem[SMEM_BYTES];
    _Float16* sFeat = (_Float16*)smem;
    _Float16* sW0t  = (_Float16*)(smem + OFF_W0);
    float*    sSm   = (float*)(smem + OFF_SM);   // [0,192)=w1, [192,195)=b1, [195,259)=b0
    float*    hT    = (float*)smem;              // overlays sFeat after WMMA: [64 n][132 m]

    const int tid  = threadIdx.x;
    const int lane = tid & 31;
    const int wave = tid >> 5;

    const int pt  = blockIdx.x * NPTS + tid;
    const int ptc = (pt < n) ? pt : (n - 1);   // clamp: WMMA needs EXEC all-1s, keep all lanes live

    // ---- stage constants: w1/b1/b0, and w0 transposed f32->f16 ----
    for (int i = tid; i < 192; i += NPTS) sSm[i] = w1[i];
    if (tid < 3)  sSm[192 + tid] = b1[tid];
    if (tid < 64) sSm[195 + tid] = b0[tid];
    for (int i = tid; i < KDIM * HID; i += NPTS) {
        const int k = i >> 6, nn = i & 63;
        sW0t[nn * LDS_WK + k] = (_Float16)w0[i];
    }

    // ---- phase 1: 9-plane bilinear gather -> f16 feature row in LDS ----
    const float4 pin = ((const float4*)inp)[ptc];
    const float lx = aabb[0], ly = aabb[1], lz = aabb[2];
    const float hx = aabb[3], hy = aabb[4], hz = aabb[5];
    float c4[4];
    c4[0] = (pin.x - lx) / (hx - lx) * 2.f - 1.f;
    c4[1] = (pin.y - ly) / (hy - ly) * 2.f - 1.f;
    c4[2] = (pin.z - lz) / (hz - lz) * 2.f - 1.f;
    c4[3] = pin.w * 2.f - 1.f;

    const int   Aidx[9] = {0,0,0,1,1,2, 0,0,1};
    const int   Bidx[9] = {1,2,3,2,3,3, 1,2,2};
    const int   Hh[9]   = {128,128,100,128,100,100, 128,128,128};
    const float* P[9]   = {tp0,tp1,tp2,tp3,tp4,tp5, sp0,sp1,sp2};

    int   off[9], HW[9];
    float w00[9], w01[9], w10[9], w11[9];
#pragma unroll
    for (int p = 0; p < 9; ++p) {
        const int Wp = 128, Hp = Hh[p];
        const float x = (c4[Aidx[p]] + 1.f) * 0.5f * (float)(Wp - 1);
        const float y = (c4[Bidx[p]] + 1.f) * 0.5f * (float)(Hp - 1);
        int x0 = (int)floorf(x); x0 = x0 < 0 ? 0 : (x0 > Wp - 2 ? Wp - 2 : x0);
        int y0 = (int)floorf(y); y0 = y0 < 0 ? 0 : (y0 > Hp - 2 ? Hp - 2 : y0);
        const float wx = x - (float)x0, wy = y - (float)y0;
        off[p] = y0 * Wp + x0;
        HW[p]  = Hp * Wp;
        w00[p] = (1.f - wx) * (1.f - wy);
        w01[p] = wx * (1.f - wy);
        w10[p] = (1.f - wx) * wy;
        w11[p] = wx * wy;
    }

    _Float16* frow = sFeat + tid * LDS_FK;
#pragma unroll 1
    for (int f = 0; f < FEAT; f += 2) {
        float aT0 = 0.f, aS0 = 0.f, aT1 = 0.f, aS1 = 0.f;
#pragma unroll
        for (int p = 0; p < 9; ++p) {
            const float* b0p = P[p] + (size_t)f * HW[p] + off[p];
            const float* b1p = b0p + HW[p];
            const float v0 = b0p[0]*w00[p] + b0p[1]*w01[p] + b0p[128]*w10[p] + b0p[129]*w11[p];
            const float v1 = b1p[0]*w00[p] + b1p[1]*w01[p] + b1p[128]*w10[p] + b1p[129]*w11[p];
            if (p < 6) { aT0 += v0; aT1 += v1; } else { aS0 += v0; aS1 += v1; }
        }
        *(unsigned*)(frow + f)        = pk2(aT0, aT1);   // 4B-aligned: f even
        *(unsigned*)(frow + FEAT + f) = pk2(aS0, aS1);
    }
    __syncthreads();

    // ---- phase 2: feat[128x128] @ w0[128x64] via v_wmma_f32_16x16x32_f16 ----
    // each wave: rows [wave*32, wave*32+32) -> 2 row-tiles x 4 col-tiles x 4 K-steps
    const int rowBase = wave * 32;
    v8f acc[8] = {};    // [rt*4 + ct]
#pragma unroll
    for (int ks = 0; ks < 4; ++ks) {
        const v16h a0 = ldA(sFeat, rowBase,      ks * 32, lane);
        const v16h a1 = ldA(sFeat, rowBase + 16, ks * 32, lane);
#pragma unroll
        for (int ct = 0; ct < 4; ++ct) {
            const v16h b = ldB(sW0t, ks * 32, ct * 16, lane);
            acc[ct]     = __builtin_amdgcn_wmma_f32_16x16x32_f16(
                              false, a0, false, b, (short)0, acc[ct],     false, false);
            acc[4 + ct] = __builtin_amdgcn_wmma_f32_16x16x32_f16(
                              false, a1, false, b, (short)0, acc[4 + ct], false, false);
        }
    }
    __syncthreads();   // all waves done reading sFeat -> safe to overlay with hT

    // ---- write h transposed: hT[n][m] = relu(acc + b0[n]) ----
    // D layout: VGPR r -> M = tileBase + r (lanes 0-15) or tileBase + r + 8 (lanes 16-31),
    // so each lane's 8 values are 8 consecutive m -> two float4 stores per tile.
    const float* sB0 = sSm + 195;
#pragma unroll
    for (int rt = 0; rt < 2; ++rt) {
#pragma unroll
        for (int ct = 0; ct < 4; ++ct) {
            const v8f a = acc[rt * 4 + ct];
            const int nn = ct * 16 + (lane & 15);
            const float bb = sB0[nn];
            const int mBase = rowBase + rt * 16 + ((lane < 16) ? 0 : 8);
            float4 lo4, hi4;
            lo4.x = fmaxf(a[0] + bb, 0.f); lo4.y = fmaxf(a[1] + bb, 0.f);
            lo4.z = fmaxf(a[2] + bb, 0.f); lo4.w = fmaxf(a[3] + bb, 0.f);
            hi4.x = fmaxf(a[4] + bb, 0.f); hi4.y = fmaxf(a[5] + bb, 0.f);
            hi4.z = fmaxf(a[6] + bb, 0.f); hi4.w = fmaxf(a[7] + bb, 0.f);
            float* dst = hT + nn * LDS_HT + mBase;
            *(float4*)(dst)     = lo4;
            *(float4*)(dst + 4) = hi4;
        }
    }
    __syncthreads();

    // ---- phase 3: out = h @ w1 + b1 (per-thread 64x3; hT read is lane-contiguous) ----
    float o0 = sSm[192], o1 = sSm[193], o2 = sSm[194];
#pragma unroll 4
    for (int j = 0; j < HID; ++j) {
        const float hv = hT[j * LDS_HT + tid];
        o0 += hv * sSm[j * 3 + 0];
        o1 += hv * sSm[j * 3 + 1];
        o2 += hv * sSm[j * 3 + 2];
    }
    if (pt < n) {
        float* op = out + (size_t)pt * 3;
        op[0] = o0; op[1] = o1; op[2] = o2;
    }
}

extern "C" void kernel_launch(void* const* d_in, const int* in_sizes, int n_in,
                              void* d_out, int out_size, void* d_ws, size_t ws_size,
                              hipStream_t stream) {
    const float* inp  = (const float*)d_in[0];
    const float* aabb = (const float*)d_in[1];
    const float* tp0  = (const float*)d_in[2];
    const float* tp1  = (const float*)d_in[3];
    const float* tp2  = (const float*)d_in[4];
    const float* tp3  = (const float*)d_in[5];
    const float* tp4  = (const float*)d_in[6];
    const float* tp5  = (const float*)d_in[7];
    const float* sp0  = (const float*)d_in[8];
    const float* sp1  = (const float*)d_in[9];
    const float* sp2  = (const float*)d_in[10];
    const float* w0   = (const float*)d_in[11];
    const float* b0   = (const float*)d_in[12];
    const float* w1   = (const float*)d_in[13];
    const float* b1   = (const float*)d_in[14];
    float* out = (float*)d_out;

    const int n = in_sizes[0] / 4;
    const int grid = (n + NPTS - 1) / NPTS;
    kplanes_mlp_kernel<<<grid, NPTS, 0, stream>>>(inp, aabb, tp0, tp1, tp2, tp3, tp4, tp5,
                                                  sp0, sp1, sp2, w0, b0, w1, b1, out, n);
}
